// MoEGate_7825430413737
// MI455X (gfx1250) — matrix operand order repaired and checked by reading, not compiled, not run on hardware
//
#include <hip/hip_runtime.h>

typedef __attribute__((ext_vector_type(16))) _Float16 v16h;
typedef __attribute__((ext_vector_type(2)))  _Float16 v2h;
typedef __attribute__((ext_vector_type(8)))  float    v8f;

#define EMBED        2048
#define NEXP         64
#define TOPK         8
#define TOK_PER_BLK  128
#define TOK_PER_WAVE 32
#define ROW_STRIDE   65   // 64 + 1 pad -> no LDS bank conflicts
#define WS_W16_OFF   512  // byte offset of f16 weight copy in workspace

// single v_cvt_pk_rtz_f16_f32; builtin returns __fp16 vec2 -> bit-identical cast
__device__ __forceinline__ v2h cvt2h(float lo, float hi) {
    return __builtin_bit_cast(v2h, __builtin_amdgcn_cvt_pkrtz(lo, hi));
}

// Fragment of 16 halfs (8 VGPRs)
union HFrag { v16h v; v2h h[8]; uint4 q[2]; };

__global__ void __launch_bounds__(128) moe_zero_ws_kernel(float* ws) {
    ws[threadIdx.x] = 0.0f;  // 128 floats: [0..63]=counts, [64..127]=Pi sums
}

// one-shot W f32 -> f16 (131072 elements, 8 per thread)
__global__ void __launch_bounds__(256) moe_wcvt_kernel(
    const float* __restrict__ w, _Float16* __restrict__ w16)
{
    int i = (blockIdx.x * 256 + threadIdx.x) * 8;
    const float4* p = (const float4*)(w + i);
    float4 a = p[0], b = p[1];
    union { v2h h[4]; uint4 q; } u;
    u.h[0] = cvt2h(a.x, a.y); u.h[1] = cvt2h(a.z, a.w);
    u.h[2] = cvt2h(b.x, b.y); u.h[3] = cvt2h(b.z, b.w);
    *(uint4*)(w16 + i) = u.q;
}

template <bool WF16>
__global__ void __launch_bounds__(128) moe_gate_main_kernel(
    const float* __restrict__ x, const float* __restrict__ w,
    const _Float16* __restrict__ w16,
    float* __restrict__ out_idx, float* __restrict__ out_w,
    float* __restrict__ ws_cnt, float* __restrict__ ws_pi, int ntok)
{
    __shared__ float lds_row[TOK_PER_BLK * ROW_STRIDE];
    __shared__ float lds_cnt[NEXP];

    const int tid  = threadIdx.x;
    const int wave = tid >> 5;
    const int lane = tid & 31;
    const int ln   = lane & 15;   // N / M index within half-wave
    const int lh   = lane >> 4;   // half-wave selector

    if (tid < NEXP) lds_cnt[tid] = 0.0f;

    const int blockTok = blockIdx.x * TOK_PER_BLK;
    const int waveTok  = blockTok + wave * TOK_PER_WAVE;

    const v8f vzero = {0.f,0.f,0.f,0.f,0.f,0.f,0.f,0.f};
    v8f acc[2][4];
    for (int t = 0; t < 2; ++t)
        for (int g = 0; g < 4; ++g)
            acc[t][g] = vzero;

    // A rows this lane feeds (16-bit A layout: lanes 0-15 and 16-31 both map M=0..15)
    const float* xrow0 = x + (size_t)(waveTok + ln)      * EMBED;
    const float* xrow1 = x + (size_t)(waveTok + 16 + ln) * EMBED;
    // B rows: expert = eg*16 + ln, 16 contiguous K values at kb + lh*16
    const float*    wrow[4];
    const _Float16* wrow16[4];
    for (int g = 0; g < 4; ++g) {
        wrow[g]   = w   + (size_t)(g * 16 + ln) * EMBED;
        wrow16[g] = w16 + (size_t)(g * 16 + ln) * EMBED;
    }

    for (int kb = 0; kb < EMBED; kb += 32) {
        __builtin_prefetch(xrow0 + kb + 256, 0, 1);
        __builtin_prefetch(xrow1 + kb + 256, 0, 1);

        // ---- A fragments: K = kb + lh*16 + (0..15)
        HFrag a0, a1;
        {
            const float4* p0 = (const float4*)(xrow0 + kb + lh * 16);
            const float4* p1 = (const float4*)(xrow0 + kb + lh * 16 + 8);
            float4 q0 = p0[0], q1 = p0[1], q2 = p1[0], q3 = p1[1];
            a0.h[0] = cvt2h(q0.x, q0.y); a0.h[1] = cvt2h(q0.z, q0.w);
            a0.h[2] = cvt2h(q1.x, q1.y); a0.h[3] = cvt2h(q1.z, q1.w);
            a0.h[4] = cvt2h(q2.x, q2.y); a0.h[5] = cvt2h(q2.z, q2.w);
            a0.h[6] = cvt2h(q3.x, q3.y); a0.h[7] = cvt2h(q3.z, q3.w);
        }
        {
            const float4* p0 = (const float4*)(xrow1 + kb + lh * 16);
            const float4* p1 = (const float4*)(xrow1 + kb + lh * 16 + 8);
            float4 q0 = p0[0], q1 = p0[1], q2 = p1[0], q3 = p1[1];
            a1.h[0] = cvt2h(q0.x, q0.y); a1.h[1] = cvt2h(q0.z, q0.w);
            a1.h[2] = cvt2h(q1.x, q1.y); a1.h[3] = cvt2h(q1.z, q1.w);
            a1.h[4] = cvt2h(q2.x, q2.y); a1.h[5] = cvt2h(q2.z, q2.w);
            a1.h[6] = cvt2h(q3.x, q3.y); a1.h[7] = cvt2h(q3.z, q3.w);
        }

        // ---- B fragments per 16-expert group, 2 WMMAs each (amortize B over 2 M tiles)
        for (int g = 0; g < 4; ++g) {
            HFrag bf;
            if constexpr (WF16) {
                // pre-converted f16 weight: 16 halfs = 32B = two b128 loads, zero cvt
                const uint4* pb = (const uint4*)(wrow16[g] + kb + lh * 16);
                bf.q[0] = pb[0];
                bf.q[1] = pb[1];
            } else {
                const float4* pb = (const float4*)(wrow[g] + kb + lh * 16);
                float4 b0 = pb[0], b1 = pb[1], b2 = pb[2], b3 = pb[3];
                bf.h[0] = cvt2h(b0.x, b0.y); bf.h[1] = cvt2h(b0.z, b0.w);
                bf.h[2] = cvt2h(b1.x, b1.y); bf.h[3] = cvt2h(b1.z, b1.w);
                bf.h[4] = cvt2h(b2.x, b2.y); bf.h[5] = cvt2h(b2.z, b2.w);
                bf.h[6] = cvt2h(b3.x, b3.y); bf.h[7] = cvt2h(b3.z, b3.w);
            }

            acc[0][g] = __builtin_amdgcn_wmma_f32_16x16x32_f16(
                false, a0.v, false, bf.v, (short)0, acc[0][g], false, false);
            acc[1][g] = __builtin_amdgcn_wmma_f32_16x16x32_f16(
                false, a1.v, false, bf.v, (short)0, acc[1][g], false, false);
        }
    }

    // ---- scatter logits to LDS: C/D layout VGPR r -> M = lh*8 + r, lane ln -> N
    for (int t = 0; t < 2; ++t)
        for (int g = 0; g < 4; ++g)
            for (int r = 0; r < 8; ++r) {
                int m = wave * TOK_PER_WAVE + t * 16 + lh * 8 + r;
                lds_row[m * ROW_STRIDE + g * 16 + ln] = acc[t][g][r];
            }
    __syncthreads();

    // ---- one thread per token: softmax + top-8 ----
    {
        float* row = &lds_row[tid * ROW_STRIDE];
        float mx = row[0];
        for (int e = 1; e < NEXP; ++e) mx = fmaxf(mx, row[e]);
        float sum = 0.f;
        for (int e = 0; e < NEXP; ++e) {
            float v = __expf(row[e] - mx);
            row[e] = v;
            sum += v;
        }
        float inv = 1.f / sum;

        float bw[TOPK]; int bi[TOPK];
        for (int j = 0; j < TOPK; ++j) { bw[j] = -1.f; bi[j] = 0; }
        for (int e = 0; e < NEXP; ++e) {
            float v = row[e];
            if (v > bw[TOPK - 1]) {
                int j = TOPK - 1;
                while (j > 0 && bw[j - 1] < v) {
                    bw[j] = bw[j - 1]; bi[j] = bi[j - 1]; --j;
                }
                bw[j] = v; bi[j] = e;
            }
        }

        size_t gtok = (size_t)blockTok + tid;
        for (int j = 0; j < TOPK; ++j) {
            out_idx[gtok * TOPK + j] = (float)bi[j];
            out_w[gtok * TOPK + j]   = bw[j] * inv;
            atomicAdd(&lds_cnt[bi[j]], 1.0f);
        }
        // normalize in place so per-expert column sums give Pi partials
        for (int e = 0; e < NEXP; ++e) row[e] *= inv;
    }
    __syncthreads();

    // ---- block-level reduction -> global atomics ----
    if (tid < NEXP) {
        float s = 0.f;
        for (int t = 0; t < TOK_PER_BLK; ++t)
            s += lds_row[t * ROW_STRIDE + tid];
        atomicAdd(&ws_pi[tid], s);
        atomicAdd(&ws_cnt[tid], lds_cnt[tid]);
    }
}

__global__ void __launch_bounds__(64) moe_gate_finalize_kernel(
    const float* __restrict__ ws_cnt, const float* __restrict__ ws_pi,
    float* __restrict__ out_aux, int ntok)
{
    __shared__ float red[NEXP];
    int e = threadIdx.x;
    float Pi = ws_pi[e] / (float)ntok;
    float ce = ws_cnt[e] / (float)(ntok * TOPK);
    red[e] = Pi * ce * (float)NEXP;
    __syncthreads();
    if (e == 0) {
        float s = 0.f;
        for (int i = 0; i < NEXP; ++i) s += red[i];
        out_aux[0] = s * 0.01f;  // ALPHA
    }
}

extern "C" void kernel_launch(void* const* d_in, const int* in_sizes, int n_in,
                              void* d_out, int out_size, void* d_ws, size_t ws_size,
                              hipStream_t stream) {
    const float* x = (const float*)d_in[0];   // [4,8192,2048] f32
    const float* w = (const float*)d_in[1];   // [64,2048] f32
    const int ntok = in_sizes[0] / EMBED;     // 32768

    float* out     = (float*)d_out;
    float* out_idx = out;                          // [ntok,8] ids (as float)
    float* out_w   = out + (size_t)ntok * TOPK;    // [ntok,8] weights
    float* out_aux = out + (size_t)ntok * TOPK * 2;// [1] aux loss

    float*    ws  = (float*)d_ws;  // [64] counts + [64] Pi sums
    _Float16* w16 = (_Float16*)((char*)d_ws + WS_W16_OFF);

    const size_t need16 = (size_t)WS_W16_OFF + (size_t)NEXP * EMBED * sizeof(_Float16);
    const bool use16 = ws_size >= need16;

    moe_zero_ws_kernel<<<1, 128, 0, stream>>>(ws);
    if (use16) {
        moe_wcvt_kernel<<<NEXP * EMBED / (256 * 8), 256, 0, stream>>>(w, w16);
        moe_gate_main_kernel<true><<<ntok / TOK_PER_BLK, TOK_PER_BLK, 0, stream>>>(
            x, w, w16, out_idx, out_w, ws, ws + NEXP, ntok);
    } else {
        moe_gate_main_kernel<false><<<ntok / TOK_PER_BLK, TOK_PER_BLK, 0, stream>>>(
            x, w, w16, out_idx, out_w, ws, ws + NEXP, ntok);
    }
    moe_gate_finalize_kernel<<<1, 64, 0, stream>>>(ws, ws + NEXP, out_aux, ntok);
}
